// DeformableConv_90598040141939
// MI455X (gfx1250) — matrix operand order repaired and checked by reading, not compiled, not run on hardware
//
#include <hip/hip_runtime.h>
#include <hip/hip_bf16.h>

// ---------------- types for WMMA ----------------
typedef __attribute__((ext_vector_type(16))) __bf16 v16bf;
typedef __attribute__((ext_vector_type(8)))  float  v8f;

union Frag {          // 16 bf16 = 8 VGPRs = 2 x uint4
    v16bf v;
    uint4 q[2];
};

// fp32 -> bf16 bits, round-to-nearest-even
__device__ __forceinline__ unsigned short f2bf(float f) {
    unsigned u = __float_as_uint(f);
    u += 0x7FFFu + ((u >> 16) & 1u);
    return (unsigned short)(u >> 16);
}

#define BATCH 8
#define CIN   256
#define CO    256
#define HH    64
#define WW    64
#define HWSZ  4096           // 64*64
#define CK    2304           // 256*9
#define KSTEPS 72            // 2304/32

// ---------------------------------------------------------------------------
// Kernel 1: offset/mask conv (3x3, pad 1) -> om[b][27][64][64], fp32
// ---------------------------------------------------------------------------
__global__ __launch_bounds__(256)
void dcn_offs_conv(const float* __restrict__ x,
                   const float* __restrict__ w_off,
                   float* __restrict__ om) {
    int t  = blockIdx.x * 256 + threadIdx.x;      // 0 .. 32767
    int b  = t >> 12;
    int yx = t & (HWSZ - 1);
    int y  = yx >> 6;
    int xx = yx & 63;

    float acc[27];
#pragma unroll
    for (int oc = 0; oc < 27; ++oc) acc[oc] = 0.f;

    const float* xb = x + (size_t)b * CIN * HWSZ;
    for (int c = 0; c < CIN; ++c) {
        float v[9];
#pragma unroll
        for (int ty = 0; ty < 3; ++ty)
#pragma unroll
            for (int tx = 0; tx < 3; ++tx) {
                int yy = y - 1 + ty;
                int xc = xx - 1 + tx;
                bool ok = ((unsigned)yy < (unsigned)HH) & ((unsigned)xc < (unsigned)WW);
                v[ty * 3 + tx] = ok ? xb[c * HWSZ + yy * WW + xc] : 0.f;
            }
        const float* wc = w_off + c * 9;          // w_off[oc][c][ty][tx]
#pragma unroll
        for (int oc = 0; oc < 27; ++oc) {
            const float* wo = wc + (size_t)oc * CIN * 9;
            float s = acc[oc];
#pragma unroll
            for (int k = 0; k < 9; ++k) s = fmaf(wo[k], v[k], s);
            acc[oc] = s;
        }
    }
#pragma unroll
    for (int oc = 0; oc < 27; ++oc)
        om[((size_t)(b * 27 + oc)) * HWSZ + yx] = acc[oc];
}

// ---------------------------------------------------------------------------
// Kernel 2: convert w_conv (fp32 [256][2304]) -> bf16 row-major
// ---------------------------------------------------------------------------
__global__ __launch_bounds__(256)
void dcn_wconv_bf16(const float* __restrict__ w, unsigned short* __restrict__ wbf) {
    int t = blockIdx.x * 256 + threadIdx.x;
    if (t < CO * CK) wbf[t] = f2bf(w[t]);
}

// ---------------------------------------------------------------------------
// Kernel 3: fused bilinear-sample + GEMM via v_wmma_f32_16x16x32_bf16
// grid = 8 batches * 64 pixel-chunks = 512 blocks, 256 threads (8 wave32)
// block output tile: 256 (Co) x 64 (pixels); wave: 32 (Co) x 64
// Double-buffered LDS tiles: staging of step k+1 overlaps WMMA of step k.
// ---------------------------------------------------------------------------
__global__ __launch_bounds__(256)
void dcn_fused_gemm(const float* __restrict__ x,
                    const float* __restrict__ om,
                    const unsigned short* __restrict__ wbf,
                    float* __restrict__ out) {
    // Precomputed sample records: per (tap k, pixel j): 4 clamped gather
    // indices + 4 bilinear weights (sigmoid mask + OOB validity folded in).
    __shared__ __align__(16) int4   s_reci[576];              // 9.2 KB
    __shared__ __align__(16) float4 s_recw[576];              // 9.2 KB
    __shared__ __align__(16) unsigned short s_a[2][CO * 32];  // 2 x 16 KB
    __shared__ __align__(16) unsigned short s_b[2][64 * 32];  // 2 x 4 KB, [j][i]

    const int tid   = threadIdx.x;
    const int b     = blockIdx.x >> 6;
    const int nbase = (blockIdx.x & 63) * 64;                // pixel chunk start
    const int wid   = tid >> 5;
    const int lane  = tid & 31;
    const int h     = lane >> 4;                             // half-wave select
    const int r     = lane & 15;
    const int m0    = wid * 32;                              // wave's Co base

    // ---- stage 1: sample records for this pixel chunk (once per block) ----
    for (int e = tid; e < 576; e += 256) {
        int k  = e >> 6;                  // tap 0..8
        int j  = e & 63;                  // pixel in chunk
        int yx = nbase + j;
        int y  = yx >> 6;
        int xx = yx & 63;
        const float* omb = om + (size_t)b * 27 * HWSZ + yx;
        float oy = omb[(size_t)k * HWSZ];
        float ox = omb[(size_t)(9 + k) * HWSZ];
        float mm = omb[(size_t)(18 + k) * HWSZ];
        mm = 1.0f / (1.0f + __expf(-mm));

        float py = (float)(y - 1 + (k / 3)) + oy;
        float px = (float)(xx - 1 + (k % 3)) + ox;
        float y0f = floorf(py), x0f = floorf(px);
        int   y0 = (int)y0f,   x0 = (int)x0f;
        int   y1 = y0 + 1,     x1 = x0 + 1;
        float ly = py - y0f,   lx = px - x0f;
        float v00 = ((unsigned)y0 < (unsigned)HH) & ((unsigned)x0 < (unsigned)WW) ? 1.f : 0.f;
        float v01 = ((unsigned)y0 < (unsigned)HH) & ((unsigned)x1 < (unsigned)WW) ? 1.f : 0.f;
        float v10 = ((unsigned)y1 < (unsigned)HH) & ((unsigned)x0 < (unsigned)WW) ? 1.f : 0.f;
        float v11 = ((unsigned)y1 < (unsigned)HH) & ((unsigned)x1 < (unsigned)WW) ? 1.f : 0.f;
        int cy0 = min(max(y0, 0), HH - 1), cy1 = min(max(y1, 0), HH - 1);
        int cx0 = min(max(x0, 0), WW - 1), cx1 = min(max(x1, 0), WW - 1);

        int4 id; float4 wt;
        id.x = cy0 * WW + cx0;  wt.x = (1.f - ly) * (1.f - lx) * v00 * mm;
        id.y = cy0 * WW + cx1;  wt.y = (1.f - ly) * lx * v01 * mm;
        id.z = cy1 * WW + cx0;  wt.z = ly * (1.f - lx) * v10 * mm;
        id.w = cy1 * WW + cx1;  wt.w = ly * lx * v11 * mm;
        s_reci[e] = id;
        s_recw[e] = wt;
    }
    __syncthreads();

    // ---- tile staging (A: weights, B: on-the-fly bilinear im2col) ----
    auto stage = [&](int buf, int k0) {
        // A tile: thread t loads w row t, 32 bf16 = 4x b128 -> LDS
        const uint4* src = (const uint4*)(wbf + (size_t)tid * CK + k0);
        uint4* dst = (uint4*)(&s_a[buf][tid * 32]);
        dst[0] = src[0]; dst[1] = src[1]; dst[2] = src[2]; dst[3] = src[3];
        if (k0 + 32 < CK)
            __builtin_prefetch(wbf + (size_t)tid * CK + k0 + 32, 0, 3);

        // B tile: 32 ck-rows x 64 pixels, col-major by pixel in LDS
        int i  = tid & 31;                 // local ck index
        int ck = k0 + i;
        int c  = ck / 9;
        int k  = ck - c * 9;
        const float* xp = x + ((size_t)(b * CIN + c)) * HWSZ;
        int j0 = (tid >> 5) * 8;
#pragma unroll
        for (int jj = 0; jj < 8; ++jj) {
            int j = j0 + jj;
            int e = k * 64 + j;
            int4   id = s_reci[e];
            float4 wt = s_recw[e];
            float v = xp[id.x] * wt.x + xp[id.y] * wt.y
                    + xp[id.z] * wt.z + xp[id.w] * wt.w;
            s_b[buf][j * 32 + i] = f2bf(v);
        }
    };

    v8f acc[2][4];
#pragma unroll
    for (int mt = 0; mt < 2; ++mt)
#pragma unroll
        for (int nt = 0; nt < 4; ++nt) acc[mt][nt] = (v8f)(0.0f);

    stage(0, 0);
    __syncthreads();

    // ---- K loop: 2304 in steps of 32, ping-pong buffers ----
    for (int kk = 0; kk < KSTEPS; ++kk) {
        const int p = kk & 1;
        if (kk + 1 < KSTEPS) stage(p ^ 1, (kk + 1) * 32);

        // fragment loads (per ISA 16-bit A 16x32 / B 32x16 VGPR layouts)
        Frag a[2], bfr[4];
#pragma unroll
        for (int mt = 0; mt < 2; ++mt) {
            const unsigned short* ap = &s_a[p][(m0 + mt * 16 + r) * 32 + h * 8];
            a[mt].q[0] = *(const uint4*)ap;          // K = {0..7}+8h
            a[mt].q[1] = *(const uint4*)(ap + 16);   // K = {16..23}+8h
        }
#pragma unroll
        for (int nt = 0; nt < 4; ++nt) {
            const unsigned short* bp = &s_b[p][(nt * 16 + r) * 32 + h * 8];
            bfr[nt].q[0] = *(const uint4*)bp;        // K = {0..7}+8h
            bfr[nt].q[1] = *(const uint4*)(bp + 16); // K = {16..23}+8h
        }

#pragma unroll
        for (int mt = 0; mt < 2; ++mt)
#pragma unroll
            for (int nt = 0; nt < 4; ++nt)
                acc[mt][nt] = __builtin_amdgcn_wmma_f32_16x16x32_bf16(
                    false, a[mt].v, false, bfr[nt].v,
                    (short)0, acc[mt][nt], false, false);
        __syncthreads();
    }

    // ---- epilogue: D layout -> out[b][co][yx] ----
    float* outb = out + (size_t)b * CO * HWSZ;
#pragma unroll
    for (int mt = 0; mt < 2; ++mt)
#pragma unroll
        for (int nt = 0; nt < 4; ++nt) {
            int yx = nbase + nt * 16 + r;
#pragma unroll
            for (int e = 0; e < 8; ++e) {
                int co = m0 + mt * 16 + e + 8 * h;
                outb[(size_t)co * HWSZ + yx] = acc[mt][nt][e];
            }
        }
}

// ---------------------------------------------------------------------------
extern "C" void kernel_launch(void* const* d_in, const int* in_sizes, int n_in,
                              void* d_out, int out_size, void* d_ws, size_t ws_size,
                              hipStream_t stream) {
    const float* x      = (const float*)d_in[0];   // (8,256,64,64)
    const float* w_off  = (const float*)d_in[1];   // (27,256,3,3)
    const float* w_conv = (const float*)d_in[2];   // (256,256,3,3)
    float* out = (float*)d_out;                    // (8,256,64,64)

    // workspace layout: om fp32 (3,538,944 B) | wbf bf16 (1,179,648 B)
    float* om = (float*)d_ws;
    unsigned short* wbf = (unsigned short*)((char*)d_ws + (size_t)BATCH * 27 * HWSZ * 4);

    dcn_offs_conv<<<(BATCH * HWSZ) / 256, 256, 0, stream>>>(x, w_off, om);
    dcn_wconv_bf16<<<(CO * CK + 255) / 256, 256, 0, stream>>>(w_conv, wbf);
    dcn_fused_gemm<<<BATCH * (HWSZ / 64), 256, 0, stream>>>(x, om, wbf, out);
}
